// MLPProjectionFilter_56118042689660
// MI455X (gfx1250) — matrix operand order repaired and checked by reading, not compiled, not run on hardware
//
#include <hip/hip_runtime.h>
#include <hip/hip_bf16.h>

// ---------------------------------------------------------------------------
// Problem constants (mirroring the reference)
// ---------------------------------------------------------------------------
#define BATCH     32768
#define NVARS     50          // NUM_STEPS * NUM_DOF
#define NCON      394         // 2*50 + 2*49 + 2*48 + 2*50
#define NQ        52          // NVARS + 2 equality rows
#define HID       1024
#define H3DIM     256
#define OUTPAD    128         // 2*NVARS = 100, padded to 128 for WMMA N
#define KPAD1     64          // K = 50 padded to 64
#define TSTEP     0.05f
#define RHO       1.0f
#define MAXITER   15
#define EPSBN     1e-5f
#define V_MAXC    1.0f
#define A_MAXC    2.0f
#define J_MAXC    5.0f
#define P_MAXC    3.14159265358979323846f
#define THETA0C   0.0f

// ---------------------------------------------------------------------------
// Types for WMMA fragments (wave32, v_wmma_f32_16x16x32_bf16)
// ---------------------------------------------------------------------------
typedef __attribute__((ext_vector_type(16))) __bf16 bf16x16;
typedef __attribute__((ext_vector_type(8)))  float  floatx8;

struct alignas(16) U8 { unsigned short s[8]; };        // 8 x bf16 = 16 bytes
union  alignas(32) FragU { bf16x16 v; U8 h[2]; };      // 16 x bf16 = 32 bytes

__device__ __forceinline__ unsigned short f2bf(float f) {
    unsigned u = __builtin_bit_cast(unsigned, f);
    unsigned r = u + 0x7FFFu + ((u >> 16) & 1u);       // round-to-nearest-even
    return (unsigned short)(r >> 16);
}
__device__ __forceinline__ float bf2f(unsigned short h) {
    return __builtin_bit_cast(float, (unsigned)h << 16);
}
__device__ __forceinline__ float waveRedAdd(float v) {
    #pragma unroll
    for (int m = 16; m >= 1; m >>= 1) v += __shfl_xor(v, m, 32);
    return v;
}

// Async global -> LDS copy of 16 bytes (GLOBAL_LOAD_ASYNC_TO_LDS_B128,
// tracked by ASYNCcnt). LDS byte address = low 32 bits of the generic
// pointer.
__device__ __forceinline__ void async_copy_b128(void* lds_dst, const void* gsrc) {
    unsigned ldsa = (unsigned)(size_t)lds_dst;
    unsigned long long ga = (unsigned long long)(size_t)gsrc;
    asm volatile("global_load_async_to_lds_b128 %0, %1, off"
                 :: "v"(ldsa), "v"(ga) : "memory");
}
__device__ __forceinline__ void async_wait0() {
    asm volatile("s_wait_asynccnt 0" ::: "memory");
}

// ---------------------------------------------------------------------------
// 0) Build solver constants on device: A_c, b_iter, b_init, Q_inv (GJ in f64)
// ---------------------------------------------------------------------------
__global__ __launch_bounds__(256, 1)
void init_consts_kernel(float* __restrict__ gAc, float* __restrict__ gQinv,
                        float* __restrict__ gbit, float* __restrict__ gbin) {
    __shared__ float  lAc[NCON * NVARS];     // 78.8 KB
    __shared__ double M[NQ * 104];           // augmented [Q | I], 43.3 KB
    const int tid = threadIdx.x;
    const float invT  = 1.0f / TSTEP;
    const float invT2 = invT * invT;

    for (int idx = tid; idx < NCON * NVARS; idx += 256) {
        int j = idx / NVARS, c = idx % NVARS;
        float a = 0.f;
        if (j < 50)        { a = (c == j) ? 1.f : 0.f; }
        else if (j < 100)  { int i = j - 50;  a = (c == i) ? -1.f : 0.f; }
        else if (j < 149)  { int i = j - 100; a = (((c == i+1)?1.f:0.f) - ((c == i)?1.f:0.f)) * invT; }
        else if (j < 198)  { int i = j - 149; a = -(((c == i+1)?1.f:0.f) - ((c == i)?1.f:0.f)) * invT; }
        else if (j < 246)  { int i = j - 198; a = (((c==i+2)?1.f:0.f) - 2.f*((c==i+1)?1.f:0.f) + ((c==i)?1.f:0.f)) * invT2; }
        else if (j < 294)  { int i = j - 246; a = -(((c==i+2)?1.f:0.f) - 2.f*((c==i+1)?1.f:0.f) + ((c==i)?1.f:0.f)) * invT2; }
        else if (j < 344)  { int i = j - 294; a = (c <= i) ? TSTEP : 0.f; }
        else               { int i = j - 344; a = (c <= i) ? -TSTEP : 0.f; }
        lAc[idx] = a;
        gAc[idx] = a;
    }
    for (int j = tid; j < NCON; j += 256) {
        float bi;
        if      (j < 100) bi = V_MAXC;
        else if (j < 198) bi = A_MAXC;
        else if (j < 294) bi = J_MAXC;
        else              bi = P_MAXC;
        gbit[j] = bi;
        float bn = bi;
        if (j >= 294 && j < 344) bn = P_MAXC - THETA0C;
        else if (j >= 344)       bn = P_MAXC + THETA0C;
        gbin[j] = bn;
    }
    __syncthreads();

    // Q = [[I + rho*Ac^T Ac, Aeq^T],[Aeq, 0]] ; augmented with identity
    for (int idx = tid; idx < NQ * NQ; idx += 256) {
        int i = idx / NQ, j = idx % NQ;
        double q = 0.0;
        if (i < NVARS && j < NVARS) {
            double s = 0.0;
            for (int r = 0; r < NCON; ++r)
                s += (double)lAc[r * NVARS + i] * (double)lAc[r * NVARS + j];
            q = (double)RHO * s + ((i == j) ? 1.0 : 0.0);
        } else if (i < NVARS) {
            int e = j - NVARS;
            q = ((e == 0 && i == 0) || (e == 1 && i == NVARS - 1)) ? 1.0 : 0.0;
        } else if (j < NVARS) {
            int e = i - NVARS;
            q = ((e == 0 && j == 0) || (e == 1 && j == NVARS - 1)) ? 1.0 : 0.0;
        }
        M[i * 104 + j]      = q;
        M[i * 104 + 52 + j] = (i == j) ? 1.0 : 0.0;
    }
    __syncthreads();

    // Gauss-Jordan (no pivoting: SPD leading block + nonsingular Schur compl.)
    for (int k = 0; k < NQ; ++k) {
        double piv = M[k * 104 + k];
        __syncthreads();
        for (int c = tid; c < 104; c += 256) M[k * 104 + c] /= piv;
        __syncthreads();
        if (tid < NQ && tid != k) {
            double f = M[tid * 104 + k];
            for (int c = 0; c < 104; ++c) M[tid * 104 + c] -= f * M[k * 104 + c];
        }
        __syncthreads();
    }
    for (int idx = tid; idx < NQ * NQ; idx += 256) {
        int i = idx / NQ, j = idx % NQ;
        gQinv[idx] = (float)M[i * 104 + 52 + j];
    }
}

// ---------------------------------------------------------------------------
// 1) fp32 -> bf16 conversion (inputs / weights, with zero padding)
// ---------------------------------------------------------------------------
__global__ void convert_x_kernel(const float* __restrict__ x, unsigned short* __restrict__ xb) {
    int idx = blockIdx.x * 256 + threadIdx.x;           // over BATCH*KPAD1
    int row = idx >> 6, c = idx & 63;
    float v = (c < NVARS) ? x[row * NVARS + c] : 0.f;
    xb[idx] = f2bf(v);
}
__global__ void convert_w_kernel(const float* __restrict__ W, unsigned short* __restrict__ Wb,
                                 int K, int N, int Kp, int Np) {
    int idx = blockIdx.x * 256 + threadIdx.x;
    if (idx >= Kp * Np) return;
    int k = idx / Np, n = idx % Np;
    float v = (k < K && n < N) ? W[k * N + n] : 0.f;
    Wb[idx] = f2bf(v);
}

// ---------------------------------------------------------------------------
// 2) bf16 WMMA GEMM: C[M,N] = A[M,K] @ B[K,N] + bias; out bf16 or fp32
//    block = 256 threads (8 waves), tile 256x64, BK=32
//    wave tile 32x64: 2 A-frags x 4 B-frags = 8 WMMA per K-step
//    A tile staged via GLOBAL_LOAD_ASYNC_TO_LDS_B128 (ASYNCcnt path)
// ---------------------------------------------------------------------------
#define BM 256
#define BN 64
#define BK 32
__global__ __launch_bounds__(256, 2)
void gemm_bf16_kernel(const unsigned short* __restrict__ A,
                      const unsigned short* __restrict__ B,
                      const float* __restrict__ bias, int biasN,
                      unsigned short* __restrict__ Cbf, float* __restrict__ Cf,
                      int M, int N, int K) {
    __shared__ alignas(16) unsigned short As[BM * BK];    // row-major [256][32], 16KB
    __shared__ alignas(16) unsigned short BsT[BN * BK];   // transposed [64][32],  4KB

    const int tid  = threadIdx.x;
    const int wave = tid >> 5;
    const int lane = tid & 31;
    const int row0 = blockIdx.y * BM;
    const int col0 = blockIdx.x * BN;

    const floatx8 zero = {0.f, 0.f, 0.f, 0.f, 0.f, 0.f, 0.f, 0.f};
    floatx8 acc[2][4];
    #pragma unroll
    for (int m = 0; m < 2; ++m)
        #pragma unroll
        for (int f = 0; f < 4; ++f) acc[m][f] = zero;

    // fragment lane mapping (per CDNA5 ISA §7.12.2)
    const int fr  = lane & 15;            // A row / B col / C col within tile
    const int kbA = (lane >> 4) * 8;      // A: K-half base (0 or 8)
    const int kbB = (lane >> 4) * 8;      // B: K-half base (0 or 8)
    const int nKT = K >> 5;

    for (int kt = 0; kt < nKT; ++kt) {
        __syncthreads();                  // prev iteration done reading LDS
        // A tile (256x32 bf16 = 16KB): 4 async 16B copies per thread,
        // chunk q covers row (tid+256q)>>2, 8-short slice ((tid+256q)&3)*8
        #pragma unroll
        for (int q = 0; q < 4; ++q) {
            const int chunk = tid + 256 * q;
            const int r  = chunk >> 2;
            const int ck = (chunk & 3) * 8;
            async_copy_b128(&As[r * BK + ck],
                            A + (size_t)(row0 + r) * K + kt * BK + ck);
        }
        // B tile: thread t reads 8 shorts of row k=(t>>3), col chunk (t&7),
        // scatters transposed into BsT[n][k]
        {
            const int k = tid >> 3, nb = (tid & 7) * 8;
            U8 bv = *(const U8*)(B + (size_t)(kt * BK + k) * N + col0 + nb);
            #pragma unroll
            for (int e = 0; e < 8; ++e) BsT[(nb + e) * BK + k] = bv.s[e];
        }
        async_wait0();                    // this wave's async copies complete
        __syncthreads();                  // all waves' LDS writes visible
        if (kt + 1 < nKT) {               // global_prefetch_b8 next tiles
            __builtin_prefetch(A + (size_t)(row0 + (tid >> 2)) * K + (kt + 1) * BK, 0, 1);
            __builtin_prefetch(B + (size_t)((kt + 1) * BK + (tid >> 3)) * N + col0, 0, 1);
        }
        // A fragments for this wave (rows wave*32 .. +31)
        FragU fa[2];
        #pragma unroll
        for (int m = 0; m < 2; ++m) {
            const int base = (wave * 32 + m * 16 + fr) * BK + kbA;
            fa[m].h[0] = *(const U8*)&As[base];
            fa[m].h[1] = *(const U8*)&As[base + 16];
        }
        #pragma unroll
        for (int f = 0; f < 4; ++f) {
            FragU fb;
            const int bbase = (f * 16 + fr) * BK + kbB;
            fb.h[0] = *(const U8*)&BsT[bbase];
            fb.h[1] = *(const U8*)&BsT[bbase + 16];
            #pragma unroll
            for (int m = 0; m < 2; ++m)
                acc[m][f] = __builtin_amdgcn_wmma_f32_16x16x32_bf16(
                    false, fa[m].v, false, fb.v, (short)0, acc[m][f], false, false);
        }
    }

    // epilogue: bias + store (C layout: lane col = lane&15, rows (lane>>4)*8+i)
    const int rb = (lane >> 4) * 8;
    #pragma unroll
    for (int m = 0; m < 2; ++m) {
        #pragma unroll
        for (int f = 0; f < 4; ++f) {
            const int col = col0 + f * 16 + fr;
            const float bv = (col < biasN) ? bias[col] : 0.f;
            #pragma unroll
            for (int i = 0; i < 8; ++i) {
                const int row = row0 + wave * 32 + m * 16 + rb + i;
                float v = acc[m][f][i] + bv;
                if (Cbf) Cbf[(size_t)row * N + col] = f2bf(v);
                else     Cf [(size_t)row * N + col] = v;
            }
        }
    }
}

// ---------------------------------------------------------------------------
// 3) Deterministic batch-norm: partial column sums, then fused apply + ReLU
// ---------------------------------------------------------------------------
#define BN_CHUNKS 64
__global__ void bn_stats_kernel(const unsigned short* __restrict__ Y,
                                float* __restrict__ sumP, float* __restrict__ sqP, int N) {
    int col = blockIdx.x * 256 + threadIdx.x;
    int chunk = blockIdx.y;
    const int rows = BATCH / BN_CHUNKS;
    float s = 0.f, q = 0.f;
    for (int r = chunk * rows; r < (chunk + 1) * rows; ++r) {
        float v = bf2f(Y[(size_t)r * N + col]);
        s += v; q += v * v;
    }
    sumP[chunk * N + col] = s;
    sqP [chunk * N + col] = q;
}
__global__ void bn_apply_kernel(unsigned short* __restrict__ Y,
                                const float* __restrict__ sumP, const float* __restrict__ sqP,
                                const float* __restrict__ g, const float* __restrict__ be, int N) {
    int col = blockIdx.x * 256 + threadIdx.x;
    int chunk = blockIdx.y;
    float s = 0.f, q = 0.f;
    for (int c = 0; c < BN_CHUNKS; ++c) { s += sumP[c * N + col]; q += sqP[c * N + col]; }
    const float mean  = s / (float)BATCH;
    const float var   = q / (float)BATCH - mean * mean;
    const float scale = g[col] * rsqrtf(var + EPSBN);
    const float shift = be[col] - mean * scale;
    const int rows = BATCH / BN_CHUNKS;
    for (int r = chunk * rows; r < (chunk + 1) * rows; ++r) {
        float v = bf2f(Y[(size_t)r * N + col]) * scale + shift;
        Y[(size_t)r * N + col] = f2bf(fmaxf(v, 0.f));
    }
}

// ---------------------------------------------------------------------------
// 4) ADMM projection solver: one wave per batch row, all state LDS-resident
// ---------------------------------------------------------------------------
__global__ __launch_bounds__(256, 1)
void solver_kernel(const float* __restrict__ C4, const float* __restrict__ xin,
                   const float* __restrict__ gAc, const float* __restrict__ gQinv,
                   const float* __restrict__ gbit, const float* __restrict__ gbin,
                   float* __restrict__ outXi, float* __restrict__ outFix,
                   float* __restrict__ outPri) {
    __shared__ float sAc[NCON * NVARS];      // 78.8 KB
    __shared__ float sQ[NQ * NQ];
    __shared__ float sbit[NCON], sbin[NCON];
    __shared__ float sS[8][NCON], sRes[8][NCON];
    __shared__ float sXi[8][64], sXiN[8][64], sLam[8][64], sX[8][64], sT[8][NQ];

    const int tid  = threadIdx.x;
    const int w    = tid >> 5;
    const int lane = tid & 31;
    const int row  = blockIdx.x * 8 + w;

    for (int i = tid; i < NCON * NVARS; i += 256) sAc[i] = gAc[i];
    for (int i = tid; i < NQ * NQ; i += 256)      sQ[i]  = gQinv[i];
    for (int i = tid; i < NCON; i += 256) { sbit[i] = gbit[i]; sbin[i] = gbin[i]; }
    __syncthreads();

    for (int i = lane; i < NVARS; i += 32) {
        sXi[w][i]  = C4[(size_t)row * OUTPAD + i];
        sLam[w][i] = C4[(size_t)row * OUTPAD + NVARS + i];
        sX[w][i]   = xin[(size_t)row * NVARS + i];
    }
    __syncthreads();
    for (int j = lane; j < NCON; j += 32) {          // s0 = relu(b_init - Ac @ xi0)
        float u = 0.f;
        #pragma unroll 10
        for (int i = 0; i < NVARS; ++i) u += sAc[j * NVARS + i] * sXi[w][i];
        sS[w][j] = fmaxf(0.f, sbin[j] - u);
    }
    __syncthreads();

    float fixAcc = 0.f, priAcc = 0.f;
    for (int it = 0; it < MAXITER; ++it) {
        // (a) rhs = [lam + x + rho * Ac^T (b_it - s), 0, 0]
        for (int c = lane; c < NQ; c += 32) {
            float acc = 0.f;
            if (c < NVARS) {
                for (int j = 0; j < NCON; ++j)
                    acc += sAc[j * NVARS + c] * (sbit[j] - sS[w][j]);
                acc = sLam[w][c] + sX[w][c] + RHO * acc;
            }
            sT[w][c] = acc;
        }
        __syncthreads();
        // (b) xi_n = (Q_inv @ rhs)[:NVARS]
        for (int i = lane; i < NVARS; i += 32) {
            float a = 0.f;
            #pragma unroll 13
            for (int k = 0; k < NQ; ++k) a += sQ[i * NQ + k] * sT[w][k];
            sXiN[w][i] = a;
        }
        __syncthreads();
        // (c) u = Ac @ xi_n; s_n = relu(b_it - u); res = relu(u - b_it)
        float rn2 = 0.f, sd2 = 0.f;
        for (int j = lane; j < NCON; j += 32) {
            float u = 0.f;
            #pragma unroll 10
            for (int i = 0; i < NVARS; ++i) u += sAc[j * NVARS + i] * sXiN[w][i];
            float sn = fmaxf(0.f, sbit[j] - u);
            float r  = fmaxf(0.f, u - sbit[j]);
            rn2 += r * r;
            float d = sn - sS[w][j]; sd2 += d * d;
            sS[w][j] = sn; sRes[w][j] = r;
        }
        __syncthreads();
        // (d) lam_n = lam - rho * Ac^T res
        float ld2 = 0.f;
        for (int c = lane; c < NVARS; c += 32) {
            float a = 0.f;
            for (int j = 0; j < NCON; ++j) a += sAc[j * NVARS + c] * sRes[w][j];
            float ln = sLam[w][c] - RHO * a;
            float d = ln - sLam[w][c]; ld2 += d * d;
            sLam[w][c] = ln;
        }
        // (e) xi diff + commit
        float xd2 = 0.f;
        for (int i = lane; i < NVARS; i += 32) {
            float d = sXi[w][i] - sXiN[w][i]; xd2 += d * d;
            sXi[w][i] = sXiN[w][i];
        }
        __syncthreads();
        rn2 = waveRedAdd(rn2); sd2 = waveRedAdd(sd2);
        ld2 = waveRedAdd(ld2); xd2 = waveRedAdd(xd2);
        priAcc += sqrtf(rn2);
        fixAcc += sqrtf(xd2) + sqrtf(ld2) + sqrtf(sd2);
    }

    for (int i = lane; i < NVARS; i += 32) outXi[(size_t)row * NVARS + i] = sXi[w][i];
    if (lane == 0) {
        outFix[row] = fixAcc / (float)MAXITER;
        outPri[row] = priAcc / (float)MAXITER;
    }
}

// ---------------------------------------------------------------------------
// Host launcher
// ---------------------------------------------------------------------------
extern "C" void kernel_launch(void* const* d_in, const int* in_sizes, int n_in,
                              void* d_out, int out_size, void* d_ws, size_t ws_size,
                              hipStream_t stream) {
    (void)in_sizes; (void)n_in; (void)out_size; (void)ws_size;
    const float* x   = (const float*)d_in[0];
    const float* W1  = (const float*)d_in[1];
    const float* b1  = (const float*)d_in[2];
    const float* g1  = (const float*)d_in[3];
    const float* be1 = (const float*)d_in[4];
    const float* W2  = (const float*)d_in[5];
    const float* b2  = (const float*)d_in[6];
    const float* g2  = (const float*)d_in[7];
    const float* be2 = (const float*)d_in[8];
    const float* W3  = (const float*)d_in[9];
    const float* b3  = (const float*)d_in[10];
    const float* g3  = (const float*)d_in[11];
    const float* be3 = (const float*)d_in[12];
    const float* W4  = (const float*)d_in[13];
    const float* b4  = (const float*)d_in[14];

    // workspace carve-up
    char* w = (char*)d_ws;
    size_t off = 0;
    auto alloc = [&](size_t bytes) { char* p = w + off; off = (off + bytes + 255) & ~(size_t)255; return p; };
    unsigned short* xbf  = (unsigned short*)alloc((size_t)BATCH * KPAD1 * 2);
    unsigned short* w1b  = (unsigned short*)alloc((size_t)KPAD1 * HID * 2);
    unsigned short* w2b  = (unsigned short*)alloc((size_t)HID * HID * 2);
    unsigned short* w3b  = (unsigned short*)alloc((size_t)HID * H3DIM * 2);
    unsigned short* w4b  = (unsigned short*)alloc((size_t)H3DIM * OUTPAD * 2);
    unsigned short* actA = (unsigned short*)alloc((size_t)BATCH * HID * 2);
    unsigned short* actB = (unsigned short*)alloc((size_t)BATCH * HID * 2);
    float* c4    = (float*)alloc((size_t)BATCH * OUTPAD * 4);
    float* sumP  = (float*)alloc((size_t)BN_CHUNKS * HID * 4);
    float* sqP   = (float*)alloc((size_t)BN_CHUNKS * HID * 4);
    float* Ac    = (float*)alloc((size_t)NCON * NVARS * 4);
    float* Qinv  = (float*)alloc((size_t)NQ * NQ * 4);
    float* bitv  = (float*)alloc((size_t)NCON * 4);
    float* binv  = (float*)alloc((size_t)NCON * 4);

    float* out   = (float*)d_out;
    float* outXi = out;
    float* outFx = out + (size_t)BATCH * NVARS;
    float* outPr = out + (size_t)BATCH * NVARS + BATCH;

    // 0) constants
    init_consts_kernel<<<1, 256, 0, stream>>>(Ac, Qinv, bitv, binv);
    // 1) conversions
    convert_x_kernel<<<(BATCH * KPAD1) / 256, 256, 0, stream>>>(x, xbf);
    convert_w_kernel<<<(KPAD1 * HID + 255) / 256, 256, 0, stream>>>(W1, w1b, NVARS, HID, KPAD1, HID);
    convert_w_kernel<<<(HID * HID + 255) / 256, 256, 0, stream>>>(W2, w2b, HID, HID, HID, HID);
    convert_w_kernel<<<(HID * H3DIM + 255) / 256, 256, 0, stream>>>(W3, w3b, HID, H3DIM, HID, H3DIM);
    convert_w_kernel<<<(H3DIM * OUTPAD + 255) / 256, 256, 0, stream>>>(W4, w4b, H3DIM, 2 * NVARS, H3DIM, OUTPAD);

    // 2) layer 1: x @ W1 + b1 -> actA ; BN + ReLU in place
    gemm_bf16_kernel<<<dim3(HID / BN, BATCH / BM), 256, 0, stream>>>(
        xbf, w1b, b1, HID, actA, nullptr, BATCH, HID, KPAD1);
    bn_stats_kernel<<<dim3(HID / 256, BN_CHUNKS), 256, 0, stream>>>(actA, sumP, sqP, HID);
    bn_apply_kernel<<<dim3(HID / 256, BN_CHUNKS), 256, 0, stream>>>(actA, sumP, sqP, g1, be1, HID);

    // 3) layer 2
    gemm_bf16_kernel<<<dim3(HID / BN, BATCH / BM), 256, 0, stream>>>(
        actA, w2b, b2, HID, actB, nullptr, BATCH, HID, HID);
    bn_stats_kernel<<<dim3(HID / 256, BN_CHUNKS), 256, 0, stream>>>(actB, sumP, sqP, HID);
    bn_apply_kernel<<<dim3(HID / 256, BN_CHUNKS), 256, 0, stream>>>(actB, sumP, sqP, g2, be2, HID);

    // 4) layer 3 (actA reused)
    gemm_bf16_kernel<<<dim3(H3DIM / BN, BATCH / BM), 256, 0, stream>>>(
        actB, w3b, b3, H3DIM, actA, nullptr, BATCH, H3DIM, HID);
    bn_stats_kernel<<<dim3(H3DIM / 256, BN_CHUNKS), 256, 0, stream>>>(actA, sumP, sqP, H3DIM);
    bn_apply_kernel<<<dim3(H3DIM / 256, BN_CHUNKS), 256, 0, stream>>>(actA, sumP, sqP, g3, be3, H3DIM);

    // 5) layer 4 -> fp32 c4 (cols [0,50)=xi0, [50,100)=lam0)
    gemm_bf16_kernel<<<dim3(OUTPAD / BN, BATCH / BM), 256, 0, stream>>>(
        actA, w4b, b4, 2 * NVARS, nullptr, c4, BATCH, OUTPAD, H3DIM);

    // 6) ADMM solver: one wave per row, 8 rows per block
    solver_kernel<<<BATCH / 8, 256, 0, stream>>>(c4, x, Ac, Qinv, bitv, binv,
                                                 outXi, outFx, outPr);
}
